// CSWinAttention_68126771249362
// MI455X (gfx1250) — matrix operand order repaired and checked by reading, not compiled, not run on hardware
//
#include <hip/hip_runtime.h>
#include <hip/hip_bf16.h>
#include <stdint.h>

// ---------------------------------------------------------------------------
// CDNA5 (gfx1250) CSWin attention: bf16 WMMA pipeline
//   1) qkv GEMM  (x @ Wqkv^T + b) -> q/k/v bf16, head-major
//   2) LePE      vlepe = v + dwconv3x3(v) + b   (per stripe window)
//   3) stripe flash attention with v_wmma_f32_16x16x32_bf16
//   4) proj GEMM (attn @ Wproj^T + b) -> fp32 out
// ---------------------------------------------------------------------------

typedef __attribute__((ext_vector_type(16))) __bf16 bf16x16;
typedef __attribute__((ext_vector_type(8)))  float  f32x8;

union FragA { uint4 q[2]; bf16x16 bf; };         // 16x32 bf16 A-operand / 32x16 B-operand
union FragC { f32x8 v; float f[8]; };            // 16x16 f32 C/D

// native converts (v_cvt_*bf16*) instead of integer RNE emulation
__device__ inline uint16_t f2bf(float f) {
  __bf16 h = (__bf16)f;
  return __builtin_bit_cast(uint16_t, h);
}
__device__ inline float bf2f(uint16_t u) {
  __bf16 h = __builtin_bit_cast(__bf16, u);
  return (float)h;
}
__device__ inline uint32_t pack2bf(float a, float b) {
  union { __bf16 h[2]; uint32_t u; } x;
  x.h[0] = (__bf16)a;
  x.h[1] = (__bf16)b;
  return x.u;
}
__device__ inline f32x8 wmma_bf16(const bf16x16 a, const bf16x16 b, const f32x8 c) {
  return __builtin_amdgcn_wmma_f32_16x16x32_bf16(false, a, false, b, (short)0, c, false, false);
}

// ---------------- constants -----------------------------------------------
#define DIM     512
#define NTOK    4096          // 64*64
#define BATCH   8
#define NHEADS  8
#define HD      64
#define MTOT    32768         // BATCH * NTOK
#define BM      128
#define BN      128
#define BK      32

// ---------------------------------------------------------------------------
// Kernel 1: qkv = x @ Wqkv^T + bqkv   -> scatter into q/k/v bf16 head-major
//   x:    [32768][512] f32 ; Wqkv: [1536][512] f32 ; bqkv: [1536]
//   q/k/v: [B][head][4096][64] bf16
// ---------------------------------------------------------------------------
__global__ __launch_bounds__(256) void qkv_gemm_kernel(
    const float* __restrict__ X, const float* __restrict__ Wq,
    const float* __restrict__ bias,
    uint16_t* __restrict__ qb, uint16_t* __restrict__ kb, uint16_t* __restrict__ vb)
{
  __shared__ uint16_t As[BM][BK];
  __shared__ uint16_t Bs[BN][BK];

  const int tid  = threadIdx.x;
  const int wave = tid >> 5;
  const int lane = tid & 31;
  const int half = lane >> 4;
  const int l16  = lane & 15;
  const int bm = blockIdx.x * BM;
  const int bn = blockIdx.y * BN;
  const int mw = (wave & 3) * 32;
  const int nw = (wave >> 2) * 64;

  FragC acc[2][4];
#pragma unroll
  for (int i = 0; i < 2; ++i)
#pragma unroll
    for (int j = 0; j < 4; ++j)
#pragma unroll
      for (int e = 0; e < 8; ++e) acc[i][j].f[e] = 0.0f;

  const int row = tid >> 1;
  const int seg = (tid & 1) << 4;

  for (int k0 = 0; k0 < DIM; k0 += BK) {
    // stage A (f32 -> bf16)
    {
      const float* s = X + (size_t)(bm + row) * DIM + k0 + seg;
      uint32_t p[8];
#pragma unroll
      for (int j = 0; j < 4; ++j) {
        float4 f = *(const float4*)(s + j * 4);
        p[j*2]   = pack2bf(f.x, f.y);
        p[j*2+1] = pack2bf(f.z, f.w);
      }
      uint4* d = (uint4*)&As[row][seg];
      d[0] = make_uint4(p[0], p[1], p[2], p[3]);
      d[1] = make_uint4(p[4], p[5], p[6], p[7]);
    }
    // stage B (Wqkv rows are N, contiguous in K)
    {
      const float* s = Wq + (size_t)(bn + row) * DIM + k0 + seg;
      uint32_t p[8];
#pragma unroll
      for (int j = 0; j < 4; ++j) {
        float4 f = *(const float4*)(s + j * 4);
        p[j*2]   = pack2bf(f.x, f.y);
        p[j*2+1] = pack2bf(f.z, f.w);
      }
      uint4* d = (uint4*)&Bs[row][seg];
      d[0] = make_uint4(p[0], p[1], p[2], p[3]);
      d[1] = make_uint4(p[4], p[5], p[6], p[7]);
    }
    __syncthreads();

    FragA af[2];
#pragma unroll
    for (int ms = 0; ms < 2; ++ms) {
      const int mr = mw + ms * 16 + l16;
      af[ms].q[0] = *(const uint4*)&As[mr][half * 8];
      af[ms].q[1] = *(const uint4*)&As[mr][16 + half * 8];
    }
#pragma unroll
    for (int ns = 0; ns < 4; ++ns) {
      FragA bf;
      const int nr = nw + ns * 16 + l16;
      bf.q[0] = *(const uint4*)&Bs[nr][half * 8];
      bf.q[1] = *(const uint4*)&Bs[nr][16 + half * 8];
#pragma unroll
      for (int ms = 0; ms < 2; ++ms)
        acc[ms][ns].v = wmma_bf16(af[ms].bf, bf.bf, acc[ms][ns].v);
    }
    __syncthreads();
  }

  // epilogue: scatter to q/k/v (n in [0,1536): which = n/512, head = (n%512)/64)
#pragma unroll
  for (int ns = 0; ns < 4; ++ns) {
    const int nbase = bn + nw + ns * 16;      // 16-aligned, stays within a 64 block
    const int which = nbase / DIM;
    const int head  = (nbase % DIM) / HD;
    const int dcol  = (nbase % HD) + l16;
    const int n     = nbase + l16;
    const float bv  = bias[n];
    uint16_t* dst = (which == 0) ? qb : (which == 1) ? kb : vb;
#pragma unroll
    for (int ms = 0; ms < 2; ++ms) {
#pragma unroll
      for (int i = 0; i < 8; ++i) {
        const int m = bm + mw + ms * 16 + i + 8 * half;
        const int b = m >> 12;
        const int t = m & (NTOK - 1);
        dst[(((size_t)(b * NHEADS + head)) * NTOK + t) * HD + dcol] =
            f2bf(acc[ms][ns].f[i] + bv);
      }
    }
  }
}

// ---------------------------------------------------------------------------
// Kernel 2: LePE  vlepe = v + dwconv3x3(v, within stripe window) + bias
//   heads 0..3 : window = 8 rows (stripe) x 64 cols   (horizontal)
//   heads 4..7 : window = 64 rows x 8 cols (stripe)   (vertical)
// One thread per (b, head, token, 8-channel chunk).
// ---------------------------------------------------------------------------
__global__ __launch_bounds__(256) void lepe_kernel(
    const uint16_t* __restrict__ vb,
    const float* __restrict__ wh, const float* __restrict__ bh,
    const float* __restrict__ wv, const float* __restrict__ bv,
    uint16_t* __restrict__ vlepe)
{
  const int gid = blockIdx.x * blockDim.x + threadIdx.x;
  if (gid >= BATCH * NHEADS * NTOK * 8) return;
  const int d0  = (gid & 7) * 8;
  const int tok = (gid >> 3) & (NTOK - 1);
  const int h   = (gid >> 15) & 7;
  const int b   = gid >> 18;
  const int y = tok >> 6, x = tok & 63;

  const bool horiz = (h < 4);
  const float* w  = horiz ? wh : wv;
  const float* bs = horiz ? bh : bv;
  int ylo, yhi, xlo, xhi;
  if (horiz) { ylo = y & ~7; yhi = ylo + 8; xlo = 0;      xhi = 64; }
  else       { xlo = x & ~7; xhi = xlo + 8; ylo = 0;      yhi = 64; }

  const uint16_t* base = vb + ((size_t)(b * NHEADS + h)) * NTOK * HD;
  float acc[8];
  {
    uint4 raw = *(const uint4*)(base + (size_t)tok * HD + d0);
    const uint16_t* e = (const uint16_t*)&raw;
#pragma unroll
    for (int i = 0; i < 8; ++i) acc[i] = bf2f(e[i]) + bs[d0 + i];   // residual + bias
  }
#pragma unroll
  for (int ky = 0; ky < 3; ++ky) {
#pragma unroll
    for (int kx = 0; kx < 3; ++kx) {
      const int ny = y + ky - 1, nx = x + kx - 1;
      if (ny < ylo || ny >= yhi || nx < xlo || nx >= xhi) continue;
      uint4 raw = *(const uint4*)(base + (size_t)(ny * 64 + nx) * HD + d0);
      const uint16_t* e = (const uint16_t*)&raw;
      const float* wp = w + (ky * 3 + kx) * HD + d0;
#pragma unroll
      for (int i = 0; i < 8; ++i) acc[i] += wp[i] * bf2f(e[i]);
    }
  }
  uint16_t out[8];
#pragma unroll
  for (int i = 0; i < 8; ++i) out[i] = f2bf(acc[i]);
  *(uint4*)(vlepe + ((size_t)(b * NHEADS + h)) * NTOK * HD + (size_t)tok * HD + d0) =
      *(const uint4*)out;
}

// ---------------------------------------------------------------------------
// Kernel 3: stripe flash attention (wave-level, bf16 WMMA)
//   512 stripes of 512 tokens; each WG = 8 waves, each wave owns 16 Q rows.
//   grid = 2048 WGs : blockIdx = (((dir*8+b)*4+h4)*8+s)*4 + rowblk
// ---------------------------------------------------------------------------
__device__ inline int stripe_tok(int dir, int s, int idx) {
  if (dir == 0) return s * 512 + idx;                 // 8 rows x 64 cols
  return (idx >> 3) * 64 + s * 8 + (idx & 7);         // 64 rows x 8 cols
}

__global__ __launch_bounds__(256) void stripe_attn_kernel(
    const uint16_t* __restrict__ qb, const uint16_t* __restrict__ kb,
    const uint16_t* __restrict__ vl, uint16_t* __restrict__ attn_out)
{
  __shared__ uint16_t Ks[64][64];      // [key][d]
  __shared__ uint16_t Vs[64][64];      // [d][key]  (transposed for B-frag reads)
  __shared__ uint16_t Ps[8][16][32];   // per-wave P scratch (C-layout -> A-layout)

  const int rb     = blockIdx.x;
  const int rowblk = rb & 3;
  const int s      = (rb >> 2) & 7;
  const int h4     = (rb >> 5) & 3;
  const int b      = (rb >> 7) & 7;
  const int dir    = rb >> 10;
  const int head   = dir * 4 + h4;
  const size_t headbase = ((size_t)(b * NHEADS + head)) * NTOK * HD;

  const int tid  = threadIdx.x;
  const int wave = tid >> 5;
  const int lane = tid & 31;
  const int half = lane >> 4;
  const int l16  = lane & 15;
  const int qbase = rowblk * 128 + wave * 16;

  // Q fragments (16 rows x 64 d = two 16x32 A-frags), in registers for the whole kernel
  FragA qf[2];
  {
    const int qt = stripe_tok(dir, s, qbase + l16);
    const uint16_t* qp = qb + headbase + (size_t)qt * HD;
    qf[0].q[0] = *(const uint4*)(qp + half * 8);
    qf[0].q[1] = *(const uint4*)(qp + 16 + half * 8);
    qf[1].q[0] = *(const uint4*)(qp + 32 + half * 8);
    qf[1].q[1] = *(const uint4*)(qp + 48 + half * 8);
  }

  FragC acc[4];
  float m_st[8], l_st[8];
#pragma unroll
  for (int nt = 0; nt < 4; ++nt)
#pragma unroll
    for (int e = 0; e < 8; ++e) acc[nt].f[e] = 0.0f;
#pragma unroll
  for (int v = 0; v < 8; ++v) { m_st[v] = -1e30f; l_st[v] = 0.0f; }

  const float scale = 0.125f;          // 64^-0.5

  for (int c0 = 0; c0 < 512; c0 += 64) {
    __syncthreads();
    // stage K chunk [key][d]
    for (int i = tid; i < 512; i += 256) {
      const int key = i >> 3, dseg = (i & 7) * 8;
      const int t = stripe_tok(dir, s, c0 + key);
      *(uint4*)&Ks[key][dseg] = *(const uint4*)(kb + headbase + (size_t)t * HD + dseg);
    }
    // stage V(lepe) chunk transposed [d][key]
    for (int i = tid; i < 512; i += 256) {
      const int key = i >> 3, dseg = (i & 7) * 8;
      const int t = stripe_tok(dir, s, c0 + key);
      uint4 raw = *(const uint4*)(vl + headbase + (size_t)t * HD + dseg);
      const uint16_t* e = (const uint16_t*)&raw;
#pragma unroll
      for (int j = 0; j < 8; ++j) Vs[dseg + j][key] = e[j];
    }
    __syncthreads();

    for (int kblk = 0; kblk < 64; kblk += 32) {
      // S = Q K^T for 32 keys (two 16-wide tiles)
      FragC s0, s1;
#pragma unroll
      for (int e = 0; e < 8; ++e) { s0.f[e] = 0.0f; s1.f[e] = 0.0f; }
#pragma unroll
      for (int kc = 0; kc < 2; ++kc) {
        FragA k0, k1;
        const int key0 = kblk + l16, key1 = kblk + 16 + l16;
        k0.q[0] = *(const uint4*)&Ks[key0][kc * 32 + half * 8];
        k0.q[1] = *(const uint4*)&Ks[key0][kc * 32 + 16 + half * 8];
        k1.q[0] = *(const uint4*)&Ks[key1][kc * 32 + half * 8];
        k1.q[1] = *(const uint4*)&Ks[key1][kc * 32 + 16 + half * 8];
        s0.v = wmma_bf16(qf[kc].bf, k0.bf, s0.v);
        s1.v = wmma_bf16(qf[kc].bf, k1.bf, s1.v);
      }
      // online softmax (per row = v + 8*half; columns striped over 16-lane group)
#pragma unroll
      for (int v = 0; v < 8; ++v) {
        s0.f[v] *= scale; s1.f[v] *= scale;
        float mx = fmaxf(s0.f[v], s1.f[v]);
        mx = fmaxf(mx, __shfl_xor(mx, 1, 32));
        mx = fmaxf(mx, __shfl_xor(mx, 2, 32));
        mx = fmaxf(mx, __shfl_xor(mx, 4, 32));
        mx = fmaxf(mx, __shfl_xor(mx, 8, 32));
        const float mnew = fmaxf(m_st[v], mx);
        const float alpha = __expf(m_st[v] - mnew);
        m_st[v] = mnew;
        const float p0 = __expf(s0.f[v] - mnew);
        const float p1 = __expf(s1.f[v] - mnew);
        s0.f[v] = p0; s1.f[v] = p1;
        l_st[v] = l_st[v] * alpha + p0 + p1;
#pragma unroll
        for (int nt = 0; nt < 4; ++nt) acc[nt].f[v] *= alpha;
      }
      // C-layout P -> LDS -> A-layout fragment (wave-private, DS is in-order)
#pragma unroll
      for (int v = 0; v < 8; ++v) {
        Ps[wave][v + 8 * half][l16]      = f2bf(s0.f[v]);
        Ps[wave][v + 8 * half][16 + l16] = f2bf(s1.f[v]);
      }
      asm volatile("s_wait_dscnt 0" ::: "memory");
      FragA pf;
      pf.q[0] = *(const uint4*)&Ps[wave][l16][half * 8];
      pf.q[1] = *(const uint4*)&Ps[wave][l16][16 + half * 8];
      // O += P @ V  (V as 32keys x 16d B-frags from transposed LDS)
#pragma unroll
      for (int nt = 0; nt < 4; ++nt) {
        FragA vf;
        const int d = nt * 16 + l16;
        vf.q[0] = *(const uint4*)&Vs[d][kblk + half * 8];
        vf.q[1] = *(const uint4*)&Vs[d][kblk + 16 + half * 8];
        acc[nt].v = wmma_bf16(pf.bf, vf.bf, acc[nt].v);
      }
    }
  }

  // finalize: reduce row sums across the 16-lane group, normalize, store bf16
#pragma unroll
  for (int v = 0; v < 8; ++v) {
    float ls = l_st[v];
    ls += __shfl_xor(ls, 1, 32);
    ls += __shfl_xor(ls, 2, 32);
    ls += __shfl_xor(ls, 4, 32);
    ls += __shfl_xor(ls, 8, 32);
    l_st[v] = 1.0f / ls;
  }
#pragma unroll
  for (int v = 0; v < 8; ++v) {
    const int rowi = qbase + v + 8 * half;
    const int tok = stripe_tok(dir, s, rowi);
    uint16_t* op = attn_out + ((size_t)(b * NTOK + tok)) * DIM + head * HD;
#pragma unroll
    for (int nt = 0; nt < 4; ++nt)
      op[nt * 16 + l16] = f2bf(acc[nt].f[v] * l_st[v]);
  }
}

// ---------------------------------------------------------------------------
// Kernel 4: out = attn_out(bf16) @ Wproj^T + bproj  -> fp32
// ---------------------------------------------------------------------------
__global__ __launch_bounds__(256) void proj_gemm_kernel(
    const uint16_t* __restrict__ A, const float* __restrict__ Wp,
    const float* __restrict__ bias, float* __restrict__ Out)
{
  __shared__ uint16_t As[BM][BK];
  __shared__ uint16_t Bs[BN][BK];

  const int tid  = threadIdx.x;
  const int wave = tid >> 5;
  const int lane = tid & 31;
  const int half = lane >> 4;
  const int l16  = lane & 15;
  const int bm = blockIdx.x * BM;
  const int bn = blockIdx.y * BN;
  const int mw = (wave & 3) * 32;
  const int nw = (wave >> 2) * 64;

  FragC acc[2][4];
#pragma unroll
  for (int i = 0; i < 2; ++i)
#pragma unroll
    for (int j = 0; j < 4; ++j)
#pragma unroll
      for (int e = 0; e < 8; ++e) acc[i][j].f[e] = 0.0f;

  const int row = tid >> 1;
  const int seg = (tid & 1) << 4;

  for (int k0 = 0; k0 < DIM; k0 += BK) {
    {
      const uint16_t* s = A + (size_t)(bm + row) * DIM + k0 + seg;
      *(uint4*)&As[row][seg]     = *(const uint4*)s;
      *(uint4*)&As[row][seg + 8] = *(const uint4*)(s + 8);
    }
    {
      const float* s = Wp + (size_t)(bn + row) * DIM + k0 + seg;
      uint32_t p[8];
#pragma unroll
      for (int j = 0; j < 4; ++j) {
        float4 f = *(const float4*)(s + j * 4);
        p[j*2]   = pack2bf(f.x, f.y);
        p[j*2+1] = pack2bf(f.z, f.w);
      }
      uint4* d = (uint4*)&Bs[row][seg];
      d[0] = make_uint4(p[0], p[1], p[2], p[3]);
      d[1] = make_uint4(p[4], p[5], p[6], p[7]);
    }
    __syncthreads();

    FragA af[2];
#pragma unroll
    for (int ms = 0; ms < 2; ++ms) {
      const int mr = mw + ms * 16 + l16;
      af[ms].q[0] = *(const uint4*)&As[mr][half * 8];
      af[ms].q[1] = *(const uint4*)&As[mr][16 + half * 8];
    }
#pragma unroll
    for (int ns = 0; ns < 4; ++ns) {
      FragA bf;
      const int nr = nw + ns * 16 + l16;
      bf.q[0] = *(const uint4*)&Bs[nr][half * 8];
      bf.q[1] = *(const uint4*)&Bs[nr][16 + half * 8];
#pragma unroll
      for (int ms = 0; ms < 2; ++ms)
        acc[ms][ns].v = wmma_bf16(af[ms].bf, bf.bf, acc[ms][ns].v);
    }
    __syncthreads();
  }

#pragma unroll
  for (int ns = 0; ns < 4; ++ns) {
    const int n = bn + nw + ns * 16 + l16;
    const float bv = bias[n];
#pragma unroll
    for (int ms = 0; ms < 2; ++ms) {
#pragma unroll
      for (int i = 0; i < 8; ++i) {
        const int m = bm + mw + ms * 16 + i + 8 * half;
        Out[(size_t)m * DIM + n] = acc[ms][ns].f[i] + bv;
      }
    }
  }
}

// ---------------------------------------------------------------------------
extern "C" void kernel_launch(void* const* d_in, const int* in_sizes, int n_in,
                              void* d_out, int out_size, void* d_ws, size_t ws_size,
                              hipStream_t stream) {
  const float* x      = (const float*)d_in[0];
  const float* Wqkv   = (const float*)d_in[1];
  const float* bqkv   = (const float*)d_in[2];
  const float* Wproj  = (const float*)d_in[3];
  const float* bproj  = (const float*)d_in[4];
  const float* lhw    = (const float*)d_in[5];
  const float* lhb    = (const float*)d_in[6];
  const float* lvw    = (const float*)d_in[7];
  const float* lvb    = (const float*)d_in[8];
  float* out = (float*)d_out;

  // carve workspace (256B aligned): q, k, v, vlepe, attn_out  (each 32 MiB bf16)
  const size_t BUF = (size_t)BATCH * NHEADS * NTOK * HD;     // 16,777,216 elements
  char* ws = (char*)d_ws;
  size_t off = 0;
  auto carve = [&](size_t elems) {
    uint16_t* p = (uint16_t*)(ws + off);
    off = (off + elems * sizeof(uint16_t) + 255) & ~(size_t)255;
    return p;
  };
  uint16_t* qb    = carve(BUF);
  uint16_t* kb    = carve(BUF);
  uint16_t* vb    = carve(BUF);
  uint16_t* vlepe = carve(BUF);
  uint16_t* aout  = carve((size_t)MTOT * DIM);
  (void)ws_size; (void)in_sizes; (void)n_in; (void)out_size;

  // 1) QKV projection: grid 256 x 12
  qkv_gemm_kernel<<<dim3(MTOT / BM, (3 * DIM) / BN), 256, 0, stream>>>(
      x, Wqkv, bqkv, qb, kb, vb);

  // 2) LePE: 8*8*4096*8 threads
  lepe_kernel<<<(BATCH * NHEADS * NTOK * 8) / 256, 256, 0, stream>>>(
      vb, lhw, lhb, lvw, lvb, vlepe);

  // 3) stripe attention: 2048 WGs x 256
  stripe_attn_kernel<<<2048, 256, 0, stream>>>(qb, kb, vlepe, aout);

  // 4) output projection: grid 256 x 4
  proj_gemm_kernel<<<dim3(MTOT / BM, DIM / BN), 256, 0, stream>>>(
      aout, Wproj, bproj, out);
}
